// MultiheadAttention_68032281969114
// MI455X (gfx1250) — matrix operand order repaired and checked by reading, not compiled, not run on hardware
//
#include <hip/hip_runtime.h>
#include <hip/hip_bf16.h>

typedef __attribute__((ext_vector_type(16))) _Float16 v16h;
typedef __attribute__((ext_vector_type(8)))  float    v8f;
typedef __attribute__((ext_vector_type(4)))  unsigned int v4u;

struct U32x8 { v4u a, b; };

static __device__ __forceinline__ v16h wmma_ldA(const _Float16* p0, const _Float16* p1) {
  // A fragment: elements 0..7 = K..K+7 (from p0), elements 8..15 = K+16..K+23 (from p1)
  U32x8 t;
  t.a = *(const v4u*)p0;
  t.b = *(const v4u*)p1;
  return __builtin_bit_cast(v16h, t);
}

static __device__ __forceinline__ v16h wmma_ldB(const _Float16* p) {
  // B fragment: elements 0..7 = K..K+7 (from p), elements 8..15 = K+16..K+23 (from p+16)
  U32x8 t;
  t.a = *(const v4u*)p;
  t.b = *(const v4u*)(p + 16);
  return __builtin_bit_cast(v16h, t);
}

static __device__ __forceinline__ v8f wmma32(v16h a, v16h b, v8f c) {
  return __builtin_amdgcn_wmma_f32_16x16x32_f16(false, a, false, b, (short)0, c, false, false);
}

// ---------------------------------------------------------------- convert
__global__ void __launch_bounds__(256) cvt_f32_f16(const float* __restrict__ in,
                                                   _Float16* __restrict__ out, int n) {
  int i = blockIdx.x * 256 + threadIdx.x;
  if (i < n) out[i] = (_Float16)in[i];
}

// ---------------------------------------------------------------- GEMM core
// C[m,n] = sum_k A[m,k] * W[n,k]   (A: MxK row-major f16, W: NxK row-major f16)
// wave tile: 16 (M) x 64 (N), K looped in chunks of 32, software-pipelined 2-deep
// so next-iteration fragment loads are in flight across the current WMMAs.
static __device__ __forceinline__ void gemm_tile_16x64(
    const _Float16* __restrict__ A, const _Float16* __restrict__ Wt,
    int m0, int n0, int Kdim, v8f acc[4]) {
  const int lane = threadIdx.x & 31;
  const int r  = lane & 15;
  const int hi = lane >> 4;
  const _Float16* ap = A  + (size_t)(m0 + r) * Kdim + hi * 8;
  const _Float16* wp = Wt + (size_t)(n0 + r) * Kdim + hi * 8;

  v16h af = wmma_ldA(ap, ap + 16);
  v16h bf0 = wmma_ldB(wp);
  v16h bf1 = wmma_ldB(wp + (size_t)16 * Kdim);
  v16h bf2 = wmma_ldB(wp + (size_t)32 * Kdim);
  v16h bf3 = wmma_ldB(wp + (size_t)48 * Kdim);

  int k0 = 0;
  for (; k0 < Kdim - 32; k0 += 32) {
    const int kn = k0 + 32;
    // prefetch next stage
    v16h afn  = wmma_ldA(ap + kn, ap + kn + 16);
    v16h bfn0 = wmma_ldB(wp + kn);
    v16h bfn1 = wmma_ldB(wp + (size_t)16 * Kdim + kn);
    v16h bfn2 = wmma_ldB(wp + (size_t)32 * Kdim + kn);
    v16h bfn3 = wmma_ldB(wp + (size_t)48 * Kdim + kn);
    // current stage compute
    acc[0] = wmma32(af, bf0, acc[0]);
    acc[1] = wmma32(af, bf1, acc[1]);
    acc[2] = wmma32(af, bf2, acc[2]);
    acc[3] = wmma32(af, bf3, acc[3]);
    af = afn; bf0 = bfn0; bf1 = bfn1; bf2 = bfn2; bf3 = bfn3;
  }
  acc[0] = wmma32(af, bf0, acc[0]);
  acc[1] = wmma32(af, bf1, acc[1]);
  acc[2] = wmma32(af, bf2, acc[2]);
  acc[3] = wmma32(af, bf3, acc[3]);
}

// QKV projection: out f16, layout mode 0: (B,H,T,DH); mode 1 (V): (B,H,DH,T)
__global__ void __launch_bounds__(256) qkv_proj_kernel(
    const _Float16* __restrict__ A, const _Float16* __restrict__ Wt,
    const float* __restrict__ bias, _Float16* __restrict__ out,
    float scale, int vmode) {
  const int wave = threadIdx.x >> 5;
  const int lane = threadIdx.x & 31;
  const int m0 = (blockIdx.x * 8 + wave) * 16;
  const int n0 = blockIdx.y * 64;
  v8f acc[4] = {};
  gemm_tile_16x64(A, Wt, m0, n0, 1024, acc);
  const int r = lane & 15, hi = lane >> 4;
#pragma unroll
  for (int nt = 0; nt < 4; ++nt) {
    int n = n0 + nt * 16 + r;
    float bv = bias[n];
    int h = n >> 6, dh = n & 63;
#pragma unroll
    for (int j = 0; j < 8; ++j) {
      int m = m0 + j + 8 * hi;
      int b = m >> 11, t = m & 2047;
      float val = (acc[nt][j] + bv) * scale;
      size_t idx;
      if (vmode) idx = (((size_t)(b * 16 + h) * 64 + dh) * 2048) + t;
      else       idx = (((size_t)(b * 16 + h) * 2048 + t) * 64) + dh;
      out[idx] = (_Float16)val;
    }
  }
}

// Output projection: attn(f16, (B*L,E)) @ Wo^T + bo -> f32 d_out
__global__ void __launch_bounds__(256) out_proj_kernel(
    const _Float16* __restrict__ A, const _Float16* __restrict__ Wt,
    const float* __restrict__ bias, float* __restrict__ out) {
  const int wave = threadIdx.x >> 5;
  const int lane = threadIdx.x & 31;
  const int m0 = (blockIdx.x * 8 + wave) * 16;
  const int n0 = blockIdx.y * 64;
  v8f acc[4] = {};
  gemm_tile_16x64(A, Wt, m0, n0, 1024, acc);
  const int r = lane & 15, hi = lane >> 4;
#pragma unroll
  for (int nt = 0; nt < 4; ++nt) {
    int n = n0 + nt * 16 + r;
    float bv = bias[n];
#pragma unroll
    for (int j = 0; j < 8; ++j) {
      int m = m0 + j + 8 * hi;
      out[(size_t)m * 1024 + n] = acc[nt][j] + bv;
    }
  }
}

// ---------------------------------------------------------------- flash attention
// Q: (BH, L, DH) f16 (pre-scaled)   K: (BH, S, DH) f16   Vt: (BH, DH, S) f16
// mask: (L,S) f32   pad: (B,S) u8   out: (B, L, E) f16
__global__ void __launch_bounds__(128) flash_kernel(
    const _Float16* __restrict__ Q, const _Float16* __restrict__ K,
    const _Float16* __restrict__ Vt, const float* __restrict__ mask,
    const unsigned char* __restrict__ pad, _Float16* __restrict__ out) {
  __shared__ _Float16 lds_p[4][16][40];   // per-wave 16x32 P tile, padded stride
  const int wave = threadIdx.x >> 5;
  const int lane = threadIdx.x & 31;
  const int r = lane & 15, hi = lane >> 4;

  const int tile  = blockIdx.x * 4 + wave;
  const int ltile = tile & 127;           // L/16 = 128
  const int bh    = tile >> 7;
  const int b     = bh >> 4;              // H = 16
  const int h     = bh & 15;
  const int l0    = ltile * 16;

  const _Float16* qb = Q  + ((size_t)bh * 2048 + l0) * 64;
  const _Float16* kb = K  + (size_t)bh * 2048 * 64;
  const _Float16* vb = Vt + (size_t)bh * 64 * 2048;
  const float*    mb = mask + (size_t)l0 * 2048;
  const unsigned char* pb = pad + (size_t)b * 2048;

  v16h qf0 = wmma_ldA(qb + r * 64 + hi * 8,      qb + r * 64 + 16 + hi * 8);
  v16h qf1 = wmma_ldA(qb + r * 64 + 32 + hi * 8, qb + r * 64 + 48 + hi * 8);

  v8f acc0 = {}, acc1 = {}, acc2 = {}, acc3 = {};
  float mrow[8], lrow[8];
#pragma unroll
  for (int j = 0; j < 8; ++j) { mrow[j] = -1e30f; lrow[j] = 0.0f; }

  _Float16* pl = &lds_p[wave][0][0];

  for (int s0 = 0; s0 < 2048; s0 += 32) {
    // ---- issue ALL global loads for this iteration up front:
    // K fragments (scores), V fragments (PV), additive mask, padding mask.
    const _Float16* kp0 = kb + (size_t)(s0 + r) * 64 + hi * 8;
    const _Float16* kp1 = kb + (size_t)(s0 + 16 + r) * 64 + hi * 8;
    v16h kf00 = wmma_ldB(kp0);
    v16h kf01 = wmma_ldB(kp0 + 32);
    v16h kf10 = wmma_ldB(kp1);
    v16h kf11 = wmma_ldB(kp1 + 32);

    const _Float16* vp = vb + (size_t)r * 2048 + s0 + hi * 8;
    v16h vf0 = wmma_ldB(vp + (size_t)0  * 2048);
    v16h vf1 = wmma_ldB(vp + (size_t)16 * 2048);
    v16h vf2 = wmma_ldB(vp + (size_t)32 * 2048);
    v16h vf3 = wmma_ldB(vp + (size_t)48 * 2048);

    bool dead[2];
    float mval[2][8];
#pragma unroll
    for (int nt = 0; nt < 2; ++nt) {
      int col = s0 + nt * 16 + r;
      dead[nt] = pb[col] != 0;
      const float* mcol = mb + col;
#pragma unroll
      for (int j = 0; j < 8; ++j)
        mval[nt][j] = mcol[(size_t)(j + 8 * hi) * 2048];
    }

    // ---- scores: Q(16x64) @ K^T -> 16x32
    v8f sc0 = {}, sc1 = {};
    sc0 = wmma32(qf0, kf00, sc0);
    sc0 = wmma32(qf1, kf01, sc0);
    sc1 = wmma32(qf0, kf10, sc1);
    sc1 = wmma32(qf1, kf11, sc1);

    // ---- additive mask + padding mask
    float sv[2][8];
#pragma unroll
    for (int j = 0; j < 8; ++j) {
      sv[0][j] = dead[0] ? -__builtin_inff() : (sc0[j] + mval[0][j]);
      sv[1][j] = dead[1] ? -__builtin_inff() : (sc1[j] + mval[1][j]);
    }

    // ---- online softmax: row max (reduce over 16 N-lanes), exp, rescale
    float fac[8];
#pragma unroll
    for (int j = 0; j < 8; ++j) {
      float v = fmaxf(sv[0][j], sv[1][j]);
      v = fmaxf(v, __shfl_xor(v, 1, 16));
      v = fmaxf(v, __shfl_xor(v, 2, 16));
      v = fmaxf(v, __shfl_xor(v, 4, 16));
      v = fmaxf(v, __shfl_xor(v, 8, 16));
      float mnew = fmaxf(mrow[j], v);
      fac[j] = __expf(mrow[j] - mnew);
      mrow[j] = mnew;
      float p0 = __expf(sv[0][j] - mnew);
      float p1 = __expf(sv[1][j] - mnew);
      lrow[j] = lrow[j] * fac[j] + p0 + p1;
      pl[(j + 8 * hi) * 40 + r]      = (_Float16)p0;
      pl[(j + 8 * hi) * 40 + 16 + r] = (_Float16)p1;
    }
#pragma unroll
    for (int j = 0; j < 8; ++j) {
      acc0[j] *= fac[j]; acc1[j] *= fac[j];
      acc2[j] *= fac[j]; acc3[j] *= fac[j];
    }

    // wave-private LDS: DS ops are in-order; make ordering explicit
    asm volatile("s_wait_dscnt 0" ::: "memory");
    // ---- P A-fragment from LDS (layout transpose C->A)
    v16h pf = wmma_ldA(pl + r * 40 + hi * 8, pl + r * 40 + 16 + hi * 8);

    // ---- PV: P(16x32) @ V(32x64); V frags were loaded at iteration start
    acc0 = wmma32(pf, vf0, acc0);
    acc1 = wmma32(pf, vf1, acc1);
    acc2 = wmma32(pf, vf2, acc2);
    acc3 = wmma32(pf, vf3, acc3);
  }

  // ---- finalize: reduce row sums, normalize, store (B, L, E) f16
#pragma unroll
  for (int j = 0; j < 8; ++j) {
    float t = lrow[j];
    t += __shfl_xor(t, 1, 16);
    t += __shfl_xor(t, 2, 16);
    t += __shfl_xor(t, 4, 16);
    t += __shfl_xor(t, 8, 16);
    lrow[j] = 1.0f / t;
  }
#pragma unroll
  for (int j = 0; j < 8; ++j) {
    size_t base = ((size_t)b * 2048 + l0 + j + 8 * hi) * 1024 + (size_t)h * 64 + r;
    out[base +  0] = (_Float16)(acc0[j] * lrow[j]);
    out[base + 16] = (_Float16)(acc1[j] * lrow[j]);
    out[base + 32] = (_Float16)(acc2[j] * lrow[j]);
    out[base + 48] = (_Float16)(acc3[j] * lrow[j]);
  }
}

// ---------------------------------------------------------------- launch
extern "C" void kernel_launch(void* const* d_in, const int* in_sizes, int n_in,
                              void* d_out, int out_size, void* d_ws, size_t ws_size,
                              hipStream_t stream) {
  (void)in_sizes; (void)n_in; (void)out_size; (void)ws_size;
  const float* query = (const float*)d_in[0];
  const float* key   = (const float*)d_in[1];
  const float* value = (const float*)d_in[2];
  const float* amask = (const float*)d_in[3];
  const unsigned char* pmask = (const unsigned char*)d_in[4];  // jax bool = 1 byte
  const float* Wq = (const float*)d_in[5];
  const float* bq = (const float*)d_in[6];
  const float* Wk = (const float*)d_in[7];
  const float* bk = (const float*)d_in[8];
  const float* Wv = (const float*)d_in[9];
  const float* bv = (const float*)d_in[10];
  const float* Wo = (const float*)d_in[11];
  const float* bo = (const float*)d_in[12];

  constexpr size_t M1 = 1u << 20;       // 1Mi elements
  _Float16* w = (_Float16*)d_ws;
  _Float16* qf16   = w;                 // 8M
  _Float16* kf16   = w + 8  * M1;       // 8M
  _Float16* vf16   = w + 16 * M1;       // 8M
  _Float16* wq16   = w + 24 * M1;       // 1M
  _Float16* wk16   = w + 25 * M1;
  _Float16* wv16   = w + 26 * M1;
  _Float16* wo16   = w + 27 * M1;
  _Float16* qproj  = w + 28 * M1;       // 8M  (BH, L, DH)
  _Float16* kproj  = w + 36 * M1;       // 8M  (BH, S, DH)
  _Float16* vprojT = w + 44 * M1;       // 8M  (BH, DH, S)
  _Float16* attno  = w + 52 * M1;       // 8M  (B, L, E)

  const int nAct = 8 * (int)M1;         // 8M activations each
  const int nW   = (int)M1;             // 1M weights each
  cvt_f32_f16<<<nAct / 256, 256, 0, stream>>>(query, qf16, nAct);
  cvt_f32_f16<<<nAct / 256, 256, 0, stream>>>(key,   kf16, nAct);
  cvt_f32_f16<<<nAct / 256, 256, 0, stream>>>(value, vf16, nAct);
  cvt_f32_f16<<<nW / 256, 256, 0, stream>>>(Wq, wq16, nW);
  cvt_f32_f16<<<nW / 256, 256, 0, stream>>>(Wk, wk16, nW);
  cvt_f32_f16<<<nW / 256, 256, 0, stream>>>(Wv, wv16, nW);
  cvt_f32_f16<<<nW / 256, 256, 0, stream>>>(Wo, wo16, nW);

  dim3 gg(64, 16);                      // M/16/8 waves, N/64
  const float scaling = 0.125f;         // DH^-0.5, DH=64
  qkv_proj_kernel<<<gg, 256, 0, stream>>>(qf16, wq16, bq, qproj,  scaling, 0);
  qkv_proj_kernel<<<gg, 256, 0, stream>>>(kf16, wk16, bk, kproj,  1.0f,    0);
  qkv_proj_kernel<<<gg, 256, 0, stream>>>(vf16, wv16, bv, vprojT, 1.0f,    1);

  flash_kernel<<<2048, 128, 0, stream>>>(qproj, kproj, vprojT, amask, pmask, attno);

  out_proj_kernel<<<gg, 256, 0, stream>>>(attno, wo16, bo, (float*)d_out);
}
